// DaVinciAttention_7808250544920
// MI455X (gfx1250) — matrix operand order repaired and checked by reading, not compiled, not run on hardware
//
#include <hip/hip_runtime.h>
#include <stdint.h>

// ---------------- problem constants ----------------
#define NTOK     2048
#define HID      5120
#define HD       128
#define NQH      40
#define NKVH     8
#define GRP      5            // NQH / NKVH
#define QKVF     7208         // 5120 + 2*1024 + 40
#define KOFF     5120
#define VOFF     6144
#define GOFF     7168
#define QK_SCALE 0.08838834764831845f   // 1/sqrt(128)
#define EPSF     1e-6f

typedef __bf16 bf16_t;
typedef bf16_t v16bf __attribute__((ext_vector_type(16)));
typedef float  v8f   __attribute__((ext_vector_type(8)));

static __device__ inline v8f wmma_bf16(v16bf a, v16bf b, v8f c) {
  // D(16x16 f32) = A(16x32 bf16) * B(32x16 bf16) + C
  return __builtin_amdgcn_wmma_f32_16x16x32_bf16(false, a, false, b, (short)0, c, false, false);
}

static __device__ inline v8f zero8() {
  v8f z;
#pragma unroll
  for (int i = 0; i < 8; ++i) z[i] = 0.f;
  return z;
}

// ---- CDNA5 async global->LDS copy (16B per lane), tracked by ASYNCcnt ----
static __device__ inline void async_copy16(void* lds_dst, const void* gsrc) {
  uint32_t l = (uint32_t)(uintptr_t)lds_dst;  // low 32 bits of flat LDS addr == LDS offset
  asm volatile("global_load_async_to_lds_b128 %0, %1, off" :: "v"(l), "v"(gsrc) : "memory");
}
static __device__ inline void wait_async0() {
  asm volatile("s_wait_asynccnt 0" ::: "memory");
}

// A-fragment (16x32 bf16): lane&15 selects row; half = lane>>4 selects K pattern:
// bytes [half*32,+32) of the 64B row (K = half*16 .. half*16+15 contiguous).
static __device__ inline v16bf load_afrag(const bf16_t* row64B, int half) {
  union { uint4 q[2]; v16bf v; } u;
  const uint4* p = (const uint4*)row64B;
  u.q[0] = p[half * 2];
  u.q[1] = p[half * 2 + 1];
  return u.v;
}
// B-fragment (32x16 bf16): lane&15 selects column row; K = half*16 .. half*16+15 contiguous.
static __device__ inline v16bf load_bfrag(const bf16_t* row64B, int half) {
  union { uint4 q[2]; v16bf v; } u;
  const uint4* p = (const uint4*)row64B;
  u.q[0] = p[half * 2];
  u.q[1] = p[half * 2 + 1];
  return u.v;
}

// =================== 1) token grouping by modality (deterministic) ===================
__global__ __launch_bounds__(256) void group_tokens(const int* __restrict__ mids, int n,
                                                    int* __restrict__ perm,
                                                    int* __restrict__ cnts,
                                                    int* __restrict__ offs) {
  __shared__ int cnt[256][3];
  const int t = threadIdx.x;
  cnt[t][0] = cnt[t][1] = cnt[t][2] = 0;
  __syncthreads();
  for (int i = t; i < n; i += 256) {
    int m = mids[i]; if (m < 0 || m > 2) m = 0;
    cnt[t][m]++;
  }
  __syncthreads();
  if (t == 0) {
    int run = 0;
    for (int m = 0; m < 3; ++m) {
      offs[m] = run;
      int g = 0;
      for (int th = 0; th < 256; ++th) { int c = cnt[th][m]; cnt[th][m] = run; run += c; g += c; }
      cnts[m] = g;
    }
  }
  __syncthreads();
  int cur[3] = {cnt[t][0], cnt[t][1], cnt[t][2]};
  for (int i = t; i < n; i += 256) {
    int m = mids[i]; if (m < 0 || m > 2) m = 0;
    perm[cur[m]++] = i;
  }
}

// =================== 2) pre-RMSNorm -> bf16 H ===================
__global__ __launch_bounds__(256) void prenorm(const float* __restrict__ X,
                                               const float* __restrict__ Wn,
                                               const int* __restrict__ mids,
                                               bf16_t* __restrict__ H) {
  const int token = blockIdx.x;
  int mid = mids[token]; if (mid < 0 || mid > 2) mid = 0;
  const float* x = X + (size_t)token * HID;
  const float* w = Wn + (size_t)mid * HID;
  const int tid = threadIdx.x, lane = tid & 31, wid = tid >> 5;

  float ss = 0.f;
  for (int i = tid; i < HID; i += 256) { float v = x[i]; ss += v * v; }
#pragma unroll
  for (int m = 1; m < 32; m <<= 1) ss += __shfl_xor(ss, m, 32);
  __shared__ float red[8];
  if (lane == 0) red[wid] = ss;
  __syncthreads();
  float tot = 0.f;
#pragma unroll
  for (int i = 0; i < 8; ++i) tot += red[i];
  const float r = rsqrtf(tot * (1.0f / HID) + EPSF);
  for (int i = tid; i < HID; i += 256)
    H[(size_t)token * HID + i] = (bf16_t)(x[i] * r * (w[i] + 1.f));
}

// =================== 3/6) grouped (MoE) GEMM, 128x128x32 tiles, bf16 WMMA ===================
// Out[token, :] = A[token, :] * W[modality(token)]^T ; W is fp32 row-major [OUTF x Kdim].
__global__ __launch_bounds__(256) void grouped_gemm(const bf16_t* __restrict__ A,
                                                    const float* __restrict__ W,
                                                    float* __restrict__ Out,
                                                    const int OUTF, const int Kdim,
                                                    const int* __restrict__ perm,
                                                    const int* __restrict__ cnts,
                                                    const int* __restrict__ offs) {
  __shared__ bf16_t ldsA[128][32];   // [token-row][k]
  __shared__ bf16_t ldsB[128][32];   // [out-col][k]

  const int group = blockIdx.y >> 4;
  const int mtile = blockIdx.y & 15;
  const int cnt = cnts[group];
  if (mtile * 128 >= cnt) return;
  const int off = offs[group];
  const int ntile = blockIdx.x;
  const float* Wg = W + (size_t)group * OUTF * Kdim;

  const int tid = threadIdx.x, lane = tid & 31, wid = tid >> 5;
  const int wm = wid & 3, wn = wid >> 2;          // 4x2 wave grid, wave tile 32x64
  const int half = lane >> 4, nl = lane & 15;

  const int arow = tid >> 1, aseg = tid & 1;      // 128 rows x 2 segs of 16 elems
  const int arowg = mtile * 128 + arow;
  const int atok = (arowg < cnt) ? perm[off + arowg] : -1;
  const int bng  = ntile * 128 + arow;

  v8f acc[2][4];
#pragma unroll
  for (int mi = 0; mi < 2; ++mi)
#pragma unroll
    for (int ni = 0; ni < 4; ++ni) acc[mi][ni] = zero8();

  const int nk = Kdim >> 5;
  for (int kc = 0; kc < nk; ++kc) {
    __syncthreads();
    // ---- A tile: bf16 activations, async DMA global->LDS ----
    {
      bf16_t* dst = &ldsA[arow][aseg * 16];
      if (atok >= 0) {
        const bf16_t* src = A + (size_t)atok * Kdim + kc * 32 + aseg * 16;
        async_copy16(dst, src);
        async_copy16(dst + 8, src + 8);
      } else {
        uint4 z = make_uint4(0, 0, 0, 0);
        ((uint4*)dst)[0] = z; ((uint4*)dst)[1] = z;
      }
    }
    // ---- B tile: fp32 weights -> bf16 ----
    {
      bf16_t* dst = &ldsB[arow][aseg * 16];
      if (bng < OUTF) {
        const float4* s4 = (const float4*)(Wg + (size_t)bng * Kdim + kc * 32 + aseg * 16);
#pragma unroll
        for (int q4 = 0; q4 < 4; ++q4) {
          float4 f = s4[q4];
          dst[q4 * 4 + 0] = (bf16_t)f.x; dst[q4 * 4 + 1] = (bf16_t)f.y;
          dst[q4 * 4 + 2] = (bf16_t)f.z; dst[q4 * 4 + 3] = (bf16_t)f.w;
        }
      } else {
        uint4 z = make_uint4(0, 0, 0, 0);
        ((uint4*)dst)[0] = z; ((uint4*)dst)[1] = z;
      }
    }
    wait_async0();
    __syncthreads();

    v16bf af0 = load_afrag(&ldsA[wm * 32 + nl][0], half);
    v16bf af1 = load_afrag(&ldsA[wm * 32 + 16 + nl][0], half);
#pragma unroll
    for (int ni = 0; ni < 4; ++ni) {
      v16bf bfr = load_bfrag(&ldsB[wn * 64 + ni * 16 + nl][0], half);
      acc[0][ni] = wmma_bf16(af0, bfr, acc[0][ni]);
      acc[1][ni] = wmma_bf16(af1, bfr, acc[1][ni]);
    }
  }

  int tok[2][8];
#pragma unroll
  for (int mi = 0; mi < 2; ++mi)
#pragma unroll
    for (int r = 0; r < 8; ++r) {
      int row = mtile * 128 + wm * 32 + mi * 16 + r + half * 8;
      tok[mi][r] = (row < cnt) ? perm[off + row] : -1;
    }
#pragma unroll
  for (int mi = 0; mi < 2; ++mi)
#pragma unroll
    for (int ni = 0; ni < 4; ++ni) {
      const int col = ntile * 128 + wn * 64 + ni * 16 + nl;
#pragma unroll
      for (int r = 0; r < 8; ++r)
        if (tok[mi][r] >= 0 && col < OUTF)
          Out[(size_t)tok[mi][r] * OUTF + col] = acc[mi][ni][r];
    }
}

// =================== 4) q/k RMSNorm + rotary + scale + gate sigmoid -> bf16 ===================
__global__ __launch_bounds__(256) void qk_prep(const float* __restrict__ qkv,
                                               const float* __restrict__ rope,
                                               const float* __restrict__ qnw,
                                               const float* __restrict__ knw,
                                               const int* __restrict__ mids,
                                               bf16_t* __restrict__ Q, bf16_t* __restrict__ K,
                                               bf16_t* __restrict__ V, float* __restrict__ gs) {
  const int token = blockIdx.x;
  int mid = mids[token]; if (mid < 0 || mid > 2) mid = 0;
  const float* row = qkv + (size_t)token * QKVF;
  const float* sp = rope + (size_t)token * HD;   // sin = rope[:, 0:64]
  const float* cp = sp + 64;                     // cos = rope[:, 64:128]
  const int lane = threadIdx.x & 31, wid = threadIdx.x >> 5;
  const int j = lane * 2;
  const float s0 = sp[j], s1 = sp[j + 1], c0 = cp[j], c1 = cp[j + 1];

  for (int slot = wid; slot < 56; slot += 8) {   // 40 q heads, 8 k heads, 8 v heads
    if (slot < 48) {
      const bool isq = slot < 40;
      const float* src = isq ? row + slot * HD : row + KOFF + (slot - 40) * HD;
      const float* w   = isq ? qnw + mid * HD  : knw + mid * HD;
      bf16_t* dst = isq ? (Q + ((size_t)token * NQH + slot) * HD)
                        : (K + ((size_t)token * NKVH + (slot - 40)) * HD);
      const float scale = isq ? QK_SCALE : 1.0f;
      float a0 = src[j], a1 = src[j + 1], b0 = src[j + 64], b1 = src[j + 65];
      float ss = a0 * a0 + a1 * a1 + b0 * b0 + b1 * b1;
#pragma unroll
      for (int m = 1; m < 32; m <<= 1) ss += __shfl_xor(ss, m, 32);
      const float r = rsqrtf(ss * (1.0f / HD) + EPSF);
      a0 = a0 * r * (w[j] + 1.f);      a1 = a1 * r * (w[j + 1] + 1.f);
      b0 = b0 * r * (w[j + 64] + 1.f); b1 = b1 * r * (w[j + 65] + 1.f);
      dst[j]      = (bf16_t)((a0 * c0 - b0 * s0) * scale);
      dst[j + 1]  = (bf16_t)((a1 * c1 - b1 * s1) * scale);
      dst[j + 64] = (bf16_t)((a0 * s0 + b0 * c0) * scale);
      dst[j + 65] = (bf16_t)((a1 * s1 + b1 * c1) * scale);
    } else {
      const int vh = slot - 48;
      const float* src = row + VOFF + vh * HD;
      bf16_t* dst = V + ((size_t)token * NKVH + vh) * HD;
      for (int d = lane; d < HD; d += 32) dst[d] = (bf16_t)src[d];
    }
  }
  if (threadIdx.x < NQH) {
    const float gv = row[GOFF + threadIdx.x];
    gs[(size_t)token * NQH + threadIdx.x] = 1.0f / (1.0f + __expf(-gv));
  }
}

// =================== 5) flash attention (GQA 40q/8kv, full softmax, WMMA bf16) ===============
__global__ __launch_bounds__(128) void fmha(const bf16_t* __restrict__ Q,
                                            const bf16_t* __restrict__ K,
                                            const bf16_t* __restrict__ V,
                                            const float* __restrict__ gs,
                                            bf16_t* __restrict__ O) {
  __shared__ bf16_t ldsK[32][HD];     // keys row-major (== K^T B-fragments)
  __shared__ bf16_t ldsVt[HD][32];    // V transposed: [dim][key]
  __shared__ bf16_t ldsP[4][16][32];  // per-wave P staging (C-layout -> A-layout)

  const int bx = blockIdx.x;
  const int qtile = bx & 31;
  const int g = (bx >> 5) % GRP;
  const int h = bx / (32 * GRP);
  const int qh = h * GRP + g;

  const int tid = threadIdx.x, lane = tid & 31, wid = tid >> 5;
  const int half = lane >> 4, nl = lane & 15;
  const int qbase = qtile * 64 + wid * 16;

  v16bf qa[4];
  {
    const bf16_t* qrow = Q + ((size_t)(qbase + nl) * NQH + qh) * HD;
#pragma unroll
    for (int c = 0; c < 4; ++c) qa[c] = load_afrag(qrow + c * 32, half);
  }
  v8f of[8];
#pragma unroll
  for (int f = 0; f < 8; ++f) of[f] = zero8();
  float mrow[8], lrow[8];
#pragma unroll
  for (int r = 0; r < 8; ++r) { mrow[r] = -3.0e38f; lrow[r] = 0.f; }

  const int lkey = tid >> 2, ld0 = (tid & 3) * 32;

  for (int kt = 0; kt < NTOK / 32; ++kt) {
    __syncthreads();
    {   // cooperative K / V^T tile load
      const size_t kvtok = (size_t)(kt * 32 + lkey);
      const uint4* sk = (const uint4*)(K + (kvtok * NKVH + h) * HD + ld0);
      uint4* dk = (uint4*)&ldsK[lkey][ld0];
      dk[0] = sk[0]; dk[1] = sk[1]; dk[2] = sk[2]; dk[3] = sk[3];
      union { uint4 q[4]; bf16_t b[32]; } vv;
      const uint4* sv = (const uint4*)(V + (kvtok * NKVH + h) * HD + ld0);
      vv.q[0] = sv[0]; vv.q[1] = sv[1]; vv.q[2] = sv[2]; vv.q[3] = sv[3];
#pragma unroll
      for (int jj = 0; jj < 32; ++jj) ldsVt[ld0 + jj][lkey] = vv.b[jj];
    }
    __syncthreads();

    v8f s0 = zero8(), s1 = zero8();
#pragma unroll
    for (int c = 0; c < 4; ++c) {
      v16bf b0 = load_bfrag(&ldsK[nl][c * 32], half);
      v16bf b1 = load_bfrag(&ldsK[16 + nl][c * 32], half);
      s0 = wmma_bf16(qa[c], b0, s0);
      s1 = wmma_bf16(qa[c], b1, s1);
    }

    float alpha[8], p0[8], p1[8];
#pragma unroll
    for (int r = 0; r < 8; ++r) {
      float t = fmaxf(s0[r], s1[r]);
#pragma unroll
      for (int m = 1; m < 16; m <<= 1) t = fmaxf(t, __shfl_xor(t, m, 32));
      const float mn = fmaxf(mrow[r], t);
      alpha[r] = __expf(mrow[r] - mn);
      mrow[r] = mn;
      p0[r] = __expf(s0[r] - mn);
      p1[r] = __expf(s1[r] - mn);
      float ps = p0[r] + p1[r];
#pragma unroll
      for (int m = 1; m < 16; m <<= 1) ps += __shfl_xor(ps, m, 32);
      lrow[r] = lrow[r] * alpha[r] + ps;
    }
#pragma unroll
    for (int f = 0; f < 8; ++f)
#pragma unroll
      for (int r = 0; r < 8; ++r) of[f][r] *= alpha[r];

#pragma unroll
    for (int r = 0; r < 8; ++r) {     // C-layout -> A-layout via per-wave LDS
      const int m = r + half * 8;
      ldsP[wid][m][nl]      = (bf16_t)p0[r];
      ldsP[wid][m][16 + nl] = (bf16_t)p1[r];
    }
    v16bf pa = load_afrag(&ldsP[wid][nl][0], half);
#pragma unroll
    for (int f = 0; f < 8; ++f) {
      v16bf bv = load_bfrag(&ldsVt[f * 16 + nl][0], half);
      of[f] = wmma_bf16(pa, bv, of[f]);
    }
  }

  float sc[8];
#pragma unroll
  for (int r = 0; r < 8; ++r) {
    const int tq = qbase + r + half * 8;
    sc[r] = gs[(size_t)tq * NQH + qh] / lrow[r];
  }
#pragma unroll
  for (int f = 0; f < 8; ++f)
#pragma unroll
    for (int r = 0; r < 8; ++r) {
      const int tq = qbase + r + half * 8;
      O[((size_t)tq * NQH + qh) * HD + f * 16 + nl] = (bf16_t)(of[f][r] * sc[r]);
    }
}

// =================== host launcher ===================
extern "C" void kernel_launch(void* const* d_in, const int* in_sizes, int n_in,
                              void* d_out, int out_size, void* d_ws, size_t ws_size,
                              hipStream_t stream) {
  (void)in_sizes; (void)n_in; (void)out_size; (void)ws_size;
  const float* hidden = (const float*)d_in[0];
  const float* rope   = (const float*)d_in[1];
  const float* prew   = (const float*)d_in[2];
  const float* qkvw   = (const float*)d_in[3];
  const float* qnw    = (const float*)d_in[4];
  const float* knw    = (const float*)d_in[5];
  const float* projw  = (const float*)d_in[6];
  const int*   mids   = (const int*)d_in[7];
  float* out = (float*)d_out;

  char* ws = (char*)d_ws;
  int* cnts = (int*)ws;
  int* offs = cnts + 3;
  int* perm = (int*)(ws + 256);
  size_t p = 16384;
  bf16_t* Hbf  = (bf16_t*)(ws + p); p += (size_t)NTOK * HID * 2;       // H, later reused as O
  float*  qkvb = (float*)(ws + p);  p += (size_t)NTOK * QKVF * 4;
  bf16_t* Qb   = (bf16_t*)(ws + p); p += (size_t)NTOK * NQH * HD * 2;
  bf16_t* Kb   = (bf16_t*)(ws + p); p += (size_t)NTOK * NKVH * HD * 2;
  bf16_t* Vb   = (bf16_t*)(ws + p); p += (size_t)NTOK * NKVH * HD * 2;
  float*  gsb  = (float*)(ws + p);  p += (size_t)NTOK * NQH * 4;

  hipLaunchKernelGGL(group_tokens, dim3(1), dim3(256), 0, stream, mids, NTOK, perm, cnts, offs);
  hipLaunchKernelGGL(prenorm, dim3(NTOK), dim3(256), 0, stream, hidden, prew, mids, Hbf);
  hipLaunchKernelGGL(grouped_gemm, dim3((QKVF + 127) / 128, 48), dim3(256), 0, stream,
                     Hbf, qkvw, qkvb, QKVF, HID, perm, cnts, offs);
  hipLaunchKernelGGL(qk_prep, dim3(NTOK), dim3(256), 0, stream,
                     qkvb, rope, qnw, knw, mids, Qb, Kb, Vb, gsb);
  hipLaunchKernelGGL(fmha, dim3(NKVH * GRP * (NTOK / 64)), dim3(128), 0, stream,
                     Qb, Kb, Vb, gsb, Hbf /* O aliases H */);
  hipLaunchKernelGGL(grouped_gemm, dim3(HID / 128, 48), dim3(256), 0, stream,
                     Hbf, projw, out, HID, HID, perm, cnts, offs);
}
